// SetAttention_Linear_Fast_20486994002301
// MI455X (gfx1250) — compile-verified, matrix operand order checked by reading, not run on hardware
//
#include <hip/hip_runtime.h>
#include <hip/hip_bf16.h>
#include <math.h>

// ---------------------------------------------------------------------------
// SetAttention_Linear_Fast for MI455X (gfx1250, wave32, WMMA 16x16x32 f16)
// B=1, T=2048, C=768, NH=12, hs=64, Lmin=4, S=1023 sets (+1 tail column)
// ---------------------------------------------------------------------------

#define T_LEN  2048
#define C_DIM  768
#define NHEAD  12
#define HS     64
#define NSETS  1023
#define SP     1024   // sets padded (+1 tail/garbage slot)

typedef __attribute__((ext_vector_type(16))) _Float16 v16h;
typedef __attribute__((ext_vector_type(8)))  _Float16 v8h;
typedef __attribute__((ext_vector_type(8)))  float    v8f;

union V16 { v16h v; v8h h[2]; };

__device__ __forceinline__ v8f wmma16(const V16& a, const V16& b, v8f c) {
  return __builtin_amdgcn_wmma_f32_16x16x32_f16(false, a.v, false, b.v,
                                                (short)0, c, false, false);
}

// ---------------- conversion / transpose helpers ----------------

__global__ void k_f32_to_f16(const float* __restrict__ in,
                             _Float16* __restrict__ out, int n) {
  int i = blockIdx.x * blockDim.x + threadIdx.x;
  if (i < n) out[i] = (_Float16)in[i];
}

// out[n][k] = (half) in[k][n]  for 768x768 weight (B operand wants N-major rows)
__global__ void k_transpose768(const float* __restrict__ in,
                               _Float16* __restrict__ out) {
  int i = blockIdx.x * blockDim.x + threadIdx.x;
  if (i >= C_DIM * C_DIM) return;
  int n = i / C_DIM, k = i % C_DIM;
  out[n * C_DIM + k] = (_Float16)in[k * C_DIM + n];
}

// Wks/Wvs [4096][64] -> WT [64][4096] f16
__global__ void k_transposeWS(const float* __restrict__ in,
                              _Float16* __restrict__ out) {
  int i = blockIdx.x * blockDim.x + threadIdx.x;
  if (i >= HS * HS * HS) return;
  int d = i / (HS * HS), f = i % (HS * HS);
  out[d * (HS * HS) + f] = (_Float16)in[f * HS + d];
}

// ---------------- generic 768-K GEMM: Y = act(A @ B) (+bias) ----------------
// A: [M x 768] f16 row-major; Bt: [768 x 768] f16, row n holds column n of B.
// act==1 : elu(y)+1 ; bias!=null : +bias[col]; writes f16 and/or f32 outputs.
__global__ void k_gemm768(const _Float16* __restrict__ A,
                          const _Float16* __restrict__ Bt,
                          const float* __restrict__ bias,
                          _Float16* __restrict__ outH,
                          float* __restrict__ outF,
                          int M, int act) {
  int wid  = (blockIdx.x * blockDim.x + threadIdx.x) >> 5;
  int lane = threadIdx.x & 31;
  const int tilesN = C_DIM / 16;
  int tilesM = M >> 4;
  if (wid >= tilesM * tilesN) return;
  int mt = wid / tilesN, nt = wid % tilesN;
  int r = lane & 15;
  int koff = (lane >> 4) * 8;
  const _Float16* arow = A  + (size_t)(mt * 16 + r) * C_DIM;
  const _Float16* brow = Bt + (size_t)(nt * 16 + r) * C_DIM;
  v8f acc = {};
  for (int kb = 0; kb < C_DIM / 32; ++kb) {
    int k0 = kb * 32 + koff;
    V16 a, b;
    a.h[0] = *(const v8h*)(arow + k0);
    a.h[1] = *(const v8h*)(arow + k0 + 16);
    b.h[0] = *(const v8h*)(brow + k0);
    b.h[1] = *(const v8h*)(brow + k0 + 16);
    acc = wmma16(a, b, acc);
  }
  int col = nt * 16 + (lane & 15);
  int rowHi = (lane >> 4) * 8;
  float bval = bias ? bias[col] : 0.0f;
  for (int j = 0; j < 8; ++j) {
    int row = mt * 16 + j + rowHi;
    float y = acc[j];
    if (act == 1) y = (y > 0.0f) ? (y + 1.0f) : __expf(y);   // elu(y)+1
    y += bval;
    if (outH) outH[(size_t)row * C_DIM + col] = (_Float16)y;
    if (outF) outF[(size_t)row * C_DIM + col] = y;
  }
}

// ---------------- per-position bilinear maps ----------------
// m_k[p][d] = sum_{i,j} k[p][i] v[p][j] Wks[i*64+j][d]   (p = t*12+h)
// Batched as 16-row GEMMs over K=4096 with A rows built on the fly.
// B operands (WksT/WvsT) are shared by all 8 waves of the block -> staged
// through LDS in 256-wide K chunks (8x cut in L2 traffic vs per-wave loads).
#define KB_CHUNK 8                      // 8 wmma K-steps = 256 flat-K per stage
#define SW_PITCH (KB_CHUNK * 32 + 8)    // +8 halves pad: conflict-free ds_load_b128

__global__ __launch_bounds__(256) void k_maps(
    const _Float16* __restrict__ kh, const _Float16* __restrict__ vh,
    const _Float16* __restrict__ WksT, const _Float16* __restrict__ WvsT,
    float* __restrict__ mK, float* __restrict__ mV) {
  __shared__ _Float16 sWk[64][SW_PITCH];
  __shared__ _Float16 sWv[64][SW_PITCH];
  int tid  = threadIdx.x;
  int wv   = tid >> 5;
  int lane = tid & 31;
  int r = lane & 15;
  int koff = (lane >> 4) * 8;
  int pBase = blockIdx.x * 128;           // 8 waves x 16 rows
  int p = pBase + wv * 16 + r;
  const _Float16* krow = kh + (size_t)p * HS;
  const _Float16* vrow = vh + (size_t)p * HS;

  // per-lane v chunks used to build the A operand (hoisted out of the loop)
  v8h vA = *(const v8h*)(vrow + koff);
  v8h vB = *(const v8h*)(vrow + koff + 16);
  v8h vC = *(const v8h*)(vrow + 32 + koff);
  v8h vD = *(const v8h*)(vrow + 32 + koff + 16);

  v8f zero = {};
  v8f accK[4] = {zero, zero, zero, zero};
  v8f accV[4] = {zero, zero, zero, zero};

  int srow = tid >> 2;                    // 0..63: B row staged by this thread
  int scol = (tid & 3) * 64;              // 4 threads cover 256 columns

  for (int ch = 0; ch < 128 / KB_CHUNK; ++ch) {   // 16 K-chunks
    int f0 = ch * (KB_CHUNK * 32);
    __syncthreads();
    for (int u = 0; u < 64; u += 8) {
      *(v8h*)&sWk[srow][scol + u] =
          *(const v8h*)(WksT + (size_t)srow * (HS * HS) + f0 + scol + u);
      *(v8h*)&sWv[srow][scol + u] =
          *(const v8h*)(WvsT + (size_t)srow * (HS * HS) + f0 + scol + u);
    }
    __syncthreads();
    for (int kb2 = 0; kb2 < KB_CHUNK; ++kb2) {
      int kb = ch * KB_CHUNK + kb2;
      _Float16 kval = krow[kb >> 1];
      V16 a;
      if (kb & 1) {
        for (int u = 0; u < 8; ++u) { a.h[0][u] = kval * vC[u]; a.h[1][u] = kval * vD[u]; }
      } else {
        for (int u = 0; u < 8; ++u) { a.h[0][u] = kval * vA[u]; a.h[1][u] = kval * vB[u]; }
      }
      int fo = kb2 * 32 + koff;
      for (int nt = 0; nt < 4; ++nt) {
        V16 bk, bv;
        bk.h[0] = *(const v8h*)&sWk[nt * 16 + r][fo];
        bk.h[1] = *(const v8h*)&sWk[nt * 16 + r][fo + 16];
        bv.h[0] = *(const v8h*)&sWv[nt * 16 + r][fo];
        bv.h[1] = *(const v8h*)&sWv[nt * 16 + r][fo + 16];
        accK[nt] = wmma16(a, bk, accK[nt]);
        accV[nt] = wmma16(a, bv, accV[nt]);
      }
    }
  }
  int rowHi = (lane >> 4) * 8;
  for (int nt = 0; nt < 4; ++nt)
    for (int j = 0; j < 8; ++j) {
      int prow = pBase + wv * 16 + j + rowHi;
      int d = nt * 16 + (lane & 15);
      mK[(size_t)prow * HS + d] = accK[nt][j];
      mV[(size_t)prow * HS + d] = accV[nt][j];
    }
}

// ---------------- hierarchy (map space, 64-dim) ----------------
// m layout: [t*12+h][64] == linear t*768 + hd. S layout: [s][768].

__global__ void k_level2(const float* __restrict__ mK, const float* __restrict__ mV,
                         float* __restrict__ Sk, float* __restrict__ Sv) {
  int i = blockIdx.x * blockDim.x + threadIdx.x;
  if (i >= (T_LEN / 4) * C_DIM) return;
  int s2 = i / C_DIM, hd = i % C_DIM;
  float ak = 0.0f, av = 0.0f;
  for (int u = 0; u < 4; ++u) {
    ak += mK[(size_t)(s2 * 4 + u) * C_DIM + hd];
    av += mV[(size_t)(s2 * 4 + u) * C_DIM + hd];
  }
  Sk[(size_t)s2 * C_DIM + hd] = ak;
  Sv[(size_t)s2 * C_DIM + hd] = av;
}

__global__ void k_tails(const float* __restrict__ mK, const float* __restrict__ mV,
                        const float* __restrict__ bks, const float* __restrict__ bvs,
                        float* __restrict__ Ktail, float* __restrict__ Vtail) {
  int i = blockIdx.x * blockDim.x + threadIdx.x;
  if (i >= T_LEN * C_DIM) return;
  int t = i / C_DIM, hd = i % C_DIM, d = hd & 63;
  int t0 = t & ~3;
  float ak = 0.0f, av = 0.0f;
  for (int u = 0; u <= (t & 3); ++u) {
    ak += mK[(size_t)(t0 + u) * C_DIM + hd];
    av += mV[(size_t)(t0 + u) * C_DIM + hd];
  }
  Ktail[i] = ak + bks[d];
  Vtail[i] = av + bvs[d];
}

__global__ void k_hier(float* __restrict__ Sk, float* __restrict__ Sv) {
  int hd = blockIdx.x * blockDim.x + threadIdx.x;
  if (hd >= C_DIM) return;
  int offPrev = 0, cnt = 512, off = 512;
  while (cnt >= 2) {
    int h2 = cnt >> 1;
    for (int u = 0; u < h2; ++u) {
      Sk[(size_t)(off + u) * C_DIM + hd] =
          Sk[(size_t)(offPrev + 2 * u) * C_DIM + hd] +
          Sk[(size_t)(offPrev + 2 * u + 1) * C_DIM + hd];
      Sv[(size_t)(off + u) * C_DIM + hd] =
          Sv[(size_t)(offPrev + 2 * u) * C_DIM + hd] +
          Sv[(size_t)(offPrev + 2 * u + 1) * C_DIM + hd];
    }
    offPrev = off; off += h2; cnt = h2;
  }
}

// pack: KsetH [h][1024][64] f16 (+bias); VsetTH [h][64][1024] f16 (+bias);
// ends[s] exclusive-end; slot s==1023 zeroed / INT_MAX.
__global__ void k_pack(const float* __restrict__ Sk, const float* __restrict__ Sv,
                       const float* __restrict__ bks, const float* __restrict__ bvs,
                       _Float16* __restrict__ KsetH, _Float16* __restrict__ VsetTH,
                       int* __restrict__ ends) {
  int i = blockIdx.x * blockDim.x + threadIdx.x;
  if (i >= SP * C_DIM) return;
  int s = i / C_DIM, hd = i % C_DIM, h = hd >> 6, d = hd & 63;
  if (s == NSETS) {
    KsetH[((size_t)h * SP + s) * HS + d] = (_Float16)0.0f;
    VsetTH[((size_t)h * HS + d) * SP + s] = (_Float16)0.0f;
    if (hd == 0) ends[s] = 0x7fffffff;
    return;
  }
  float kk = Sk[(size_t)s * C_DIM + hd] + bks[d];
  float vv = Sv[(size_t)s * C_DIM + hd] + bvs[d];
  KsetH[((size_t)h * SP + s) * HS + d] = (_Float16)kk;
  VsetTH[((size_t)h * HS + d) * SP + s] = (_Float16)vv;
  if (hd == 0) {
    int base = 0, cnt = 512, l = 2;
    while (s >= base + cnt) { base += cnt; cnt >>= 1; ++l; }
    ends[s] = (s - base + 1) << l;
  }
}

// ---------------- fused set-attention (1 wave / 16 t-rows / head) ----------------
__global__ __launch_bounds__(32) void k_attn(
    const _Float16* __restrict__ qh, const _Float16* __restrict__ KsetH,
    const _Float16* __restrict__ VsetTH, const float* __restrict__ Ktail,
    const float* __restrict__ Vtail, const int* __restrict__ ends,
    _Float16* __restrict__ outh) {
  __shared__ float Lg[16][SP];      // logits -> exp values (in place)
  __shared__ float rinvArr[16];
  int h  = blockIdx.x >> 7;         // / 128 t-tiles
  int tt = blockIdx.x & 127;
  int t0 = tt * 16;
  int lane = threadIdx.x;
  int r = lane & 15;
  int koff = (lane >> 4) * 8;
  int rowHi = (lane >> 4) * 8;
  const float scale = 0.125f;       // hs^-0.5

  // Q tile in A-operand layout (K = 64 -> two 32-chunks)
  const _Float16* qrow = qh + (size_t)(t0 + r) * C_DIM + h * HS;
  V16 a0, a1;
  a0.h[0] = *(const v8h*)(qrow + koff);
  a0.h[1] = *(const v8h*)(qrow + koff + 16);
  a1.h[0] = *(const v8h*)(qrow + 32 + koff);
  a1.h[1] = *(const v8h*)(qrow + 32 + koff + 16);

  // phase 1: logits over all 64 set-tiles (s = 0..1023)
  for (int st = 0; st < 64; ++st) {
    const _Float16* krow = KsetH + ((size_t)h * SP + st * 16 + r) * HS;
    V16 b0, b1;
    b0.h[0] = *(const v8h*)(krow + koff);
    b0.h[1] = *(const v8h*)(krow + koff + 16);
    b1.h[0] = *(const v8h*)(krow + 32 + koff);
    b1.h[1] = *(const v8h*)(krow + 32 + koff + 16);
    v8f acc = {};
    acc = wmma16(a0, b0, acc);
    acc = wmma16(a1, b1, acc);
    int s = st * 16 + (lane & 15);
    int e = ends[s];
    for (int j = 0; j < 8; ++j) {
      int row = j + rowHi;
      float lg = acc[j] * scale;
      if (e > t0 + row + 1) lg = -1e30f;   // causal set mask
      Lg[row][s] = lg;
    }
  }
  // ragged tail logit -> column 1023
  if (lane < 16) {
    int t = t0 + lane;
    const _Float16* q2 = qh + (size_t)t * C_DIM + h * HS;
    const float* kt = Ktail + (size_t)(t * NHEAD + h) * HS;
    float acc = 0.0f;
    for (int d = 0; d < HS; ++d) acc += (float)q2[d] * kt[d];
    Lg[lane][NSETS] = acc * scale;
  }
  __syncthreads();

  // phase 2: softmax (2 lanes per row, in-place exp)
  {
    int row = lane & 15;
    int c0 = (lane >> 4) * 512;
    float mx = -3.0e38f;
    for (int c = 0; c < 512; ++c) mx = fmaxf(mx, Lg[row][c0 + c]);
    mx = fmaxf(mx, __shfl_xor(mx, 16, 32));
    float sum = 0.0f;
    for (int c = 0; c < 512; ++c) {
      float e = __expf(Lg[row][c0 + c] - mx);
      Lg[row][c0 + c] = e;
      sum += e;
    }
    sum += __shfl_xor(sum, 16, 32);
    if ((lane >> 4) == 0) rinvArr[row] = 1.0f / sum;
  }
  __syncthreads();

  // phase 3: out = P[16x1024] @ V^T  (column 1023 of V is zero)
  v8f zero = {};
  v8f acc[4] = {zero, zero, zero, zero};
  for (int kb = 0; kb < 32; ++kb) {
    int c0 = kb * 32 + koff;
    V16 a;
    for (int u = 0; u < 8; ++u) {
      a.h[0][u] = (_Float16)Lg[r][c0 + u];
      a.h[1][u] = (_Float16)Lg[r][c0 + 16 + u];
    }
    for (int nt = 0; nt < 4; ++nt) {
      const _Float16* vrow = VsetTH + ((size_t)h * HS + nt * 16 + r) * SP + c0;
      V16 b;
      b.h[0] = *(const v8h*)(vrow);
      b.h[1] = *(const v8h*)(vrow + 16);
      acc[nt] = wmma16(a, b, acc[nt]);
    }
  }
  // epilogue: add tail term, normalize, store f16 for final GEMM
  for (int nt = 0; nt < 4; ++nt)
    for (int j = 0; j < 8; ++j) {
      int row = j + rowHi;
      int t = t0 + row;
      int d = nt * 16 + (lane & 15);
      float o = acc[nt][j] +
                Lg[row][NSETS] * Vtail[(size_t)(t * NHEAD + h) * HS + d];
      o *= rinvArr[row];
      outh[(size_t)t * C_DIM + h * HS + d] = (_Float16)o;
    }
}

// ---------------------------------------------------------------------------

extern "C" void kernel_launch(void* const* d_in, const int* in_sizes, int n_in,
                              void* d_out, int out_size, void* d_ws, size_t ws_size,
                              hipStream_t stream) {
  const float* x   = (const float*)d_in[0];
  const float* Wq  = (const float*)d_in[1];
  const float* Wk  = (const float*)d_in[2];
  const float* Wv  = (const float*)d_in[3];
  const float* Wks = (const float*)d_in[4];
  const float* bks = (const float*)d_in[5];
  const float* Wvs = (const float*)d_in[6];
  const float* bvs = (const float*)d_in[7];
  const float* Wc  = (const float*)d_in[8];
  const float* bc  = (const float*)d_in[9];
  float* out = (float*)d_out;

  char* ws = (char*)d_ws;
  size_t off = 0;
  auto alloc = [&](size_t bytes) -> void* {
    void* p = ws + off;
    off += (bytes + 255) & ~(size_t)255;
    return p;
  };
  _Float16* xh     = (_Float16*)alloc((size_t)T_LEN * C_DIM * 2);
  _Float16* WqT    = (_Float16*)alloc((size_t)C_DIM * C_DIM * 2);
  _Float16* WkT    = (_Float16*)alloc((size_t)C_DIM * C_DIM * 2);
  _Float16* WvT    = (_Float16*)alloc((size_t)C_DIM * C_DIM * 2);
  _Float16* WcT    = (_Float16*)alloc((size_t)C_DIM * C_DIM * 2);
  _Float16* WksT   = (_Float16*)alloc((size_t)HS * HS * HS * 2);
  _Float16* WvsT   = (_Float16*)alloc((size_t)HS * HS * HS * 2);
  _Float16* qh     = (_Float16*)alloc((size_t)T_LEN * C_DIM * 2);
  _Float16* kh     = (_Float16*)alloc((size_t)T_LEN * C_DIM * 2);
  _Float16* vh     = (_Float16*)alloc((size_t)T_LEN * C_DIM * 2);
  float*    mK     = (float*)alloc((size_t)T_LEN * C_DIM * 4);
  float*    mV     = (float*)alloc((size_t)T_LEN * C_DIM * 4);
  float*    Sk     = (float*)alloc((size_t)SP * C_DIM * 4);
  float*    Sv     = (float*)alloc((size_t)SP * C_DIM * 4);
  float*    Ktail  = (float*)alloc((size_t)T_LEN * C_DIM * 4);
  float*    Vtail  = (float*)alloc((size_t)T_LEN * C_DIM * 4);
  _Float16* KsetH  = (_Float16*)alloc((size_t)NHEAD * SP * HS * 2);
  _Float16* VsetTH = (_Float16*)alloc((size_t)NHEAD * HS * SP * 2);
  int*      ends   = (int*)alloc((size_t)SP * 4);
  _Float16* outh   = (_Float16*)alloc((size_t)T_LEN * C_DIM * 2);

  // 1) conversions / transposes
  {
    int n = T_LEN * C_DIM;
    k_f32_to_f16<<<(n + 255) / 256, 256, 0, stream>>>(x, xh, n);
  }
  k_transpose768<<<(C_DIM * C_DIM + 255) / 256, 256, 0, stream>>>(Wq, WqT);
  k_transpose768<<<(C_DIM * C_DIM + 255) / 256, 256, 0, stream>>>(Wk, WkT);
  k_transpose768<<<(C_DIM * C_DIM + 255) / 256, 256, 0, stream>>>(Wv, WvT);
  k_transpose768<<<(C_DIM * C_DIM + 255) / 256, 256, 0, stream>>>(Wc, WcT);
  k_transposeWS<<<(HS * HS * HS + 255) / 256, 256, 0, stream>>>(Wks, WksT);
  k_transposeWS<<<(HS * HS * HS + 255) / 256, 256, 0, stream>>>(Wvs, WvsT);

  // 2) projections (WMMA): q,k with elu+1, v plain
  {
    int waves = (T_LEN / 16) * (C_DIM / 16);
    int blocks = (waves * 32 + 255) / 256;
    k_gemm768<<<blocks, 256, 0, stream>>>(xh, WqT, nullptr, qh, nullptr, T_LEN, 1);
    k_gemm768<<<blocks, 256, 0, stream>>>(xh, WkT, nullptr, kh, nullptr, T_LEN, 1);
    k_gemm768<<<blocks, 256, 0, stream>>>(xh, WvT, nullptr, vh, nullptr, T_LEN, 0);
  }

  // 3) per-position bilinear maps (WMMA, dominant FLOPs, LDS-staged B)
  {
    int blocks = (T_LEN * NHEAD) / 128;         // 192 blocks of 8 waves
    k_maps<<<blocks, 256, 0, stream>>>(kh, vh, WksT, WvsT, mK, mV);
  }

  // 4) hierarchy + tails + pack
  {
    int n = (T_LEN / 4) * C_DIM;
    k_level2<<<(n + 255) / 256, 256, 0, stream>>>(mK, mV, Sk, Sv);
  }
  {
    int n = T_LEN * C_DIM;
    k_tails<<<(n + 255) / 256, 256, 0, stream>>>(mK, mV, bks, bvs, Ktail, Vtail);
  }
  k_hier<<<(C_DIM + 255) / 256, 256, 0, stream>>>(Sk, Sv);
  {
    int n = SP * C_DIM;
    k_pack<<<(n + 255) / 256, 256, 0, stream>>>(Sk, Sv, bks, bvs, KsetH, VsetTH, ends);
  }

  // 5) fused set attention (WMMA logits + WMMA P@V)
  k_attn<<<NHEAD * (T_LEN / 16), 32, 0, stream>>>(qh, KsetH, VsetTH, Ktail,
                                                  Vtail, ends, outh);

  // 6) final projection (WMMA) + bias, f32 output
  {
    int waves = (T_LEN / 16) * (C_DIM / 16);
    int blocks = (waves * 32 + 255) / 256;
    k_gemm768<<<blocks, 256, 0, stream>>>(outh, WcT, bc, nullptr, out, T_LEN, 0);
  }
}
